// KRVCFlashAttention_45131516346804
// MI455X (gfx1250) — compile-verified, matrix operand order checked by reading, not run on hardware
//
#include <hip/hip_runtime.h>
#include <hip/hip_bf16.h>
#include <stdint.h>

// ---------------------------------------------------------------------------
// MHA forward for MI455X (gfx1250, wave32, WMMA bf16 16x16x32, f32 accum)
// B=2, N=2048, C=1024, H=16, D=64, SCALE = 1/8 (folded into Q)
// GEMMs: double-buffered GLOBAL_LOAD_ASYNC_TO_LDS_B128 staging (ASYNCcnt)
// Attention: TDM TENSOR_LOAD_TO_LDS for K tiles (TENSORcnt) + WMMA flash attn
// ---------------------------------------------------------------------------

typedef __attribute__((ext_vector_type(16))) __bf16        v16bf;
typedef __attribute__((ext_vector_type(8)))  float         v8f;
typedef __attribute__((ext_vector_type(4)))  unsigned int  u32x4;
typedef __attribute__((ext_vector_type(8)))  int           i32x8;
typedef __attribute__((ext_vector_type(4)))  int           i32x4;

#define AS1 __attribute__((address_space(1)))
#define AS3 __attribute__((address_space(3)))

#define SEQ_N   2048
#define EMB_C   1024
#define HEADS   16
#define HDIM    64
#define BATCH   2
#define QK_SCALE 0.125f

#if __has_builtin(__builtin_amdgcn_global_load_async_to_lds_b128)
#define HAVE_ASYNC 1
#else
#define HAVE_ASYNC 0
#endif

#if __has_builtin(__builtin_amdgcn_tensor_load_to_lds) && \
    __has_builtin(__builtin_amdgcn_s_wait_tensorcnt)
#define HAVE_TDM 1
#else
#define HAVE_TDM 0
#endif

#if HAVE_ASYNC
#if __has_builtin(__builtin_amdgcn_s_wait_asynccnt)
#define WAIT_ASYNC(n) __builtin_amdgcn_s_wait_asynccnt(n)
#else
#define WAIT_ASYNC(n) asm volatile("s_wait_asynccnt %0" ::"i"(n) : "memory")
#endif
#endif

struct FragBF {
    union { v16bf v; u32x4 u[2]; };
};

// Load a 16x32 bf16 A/B-style fragment from a row-major (along K) tile.
// lane l: matrix row = (base row) + (l & 15); element j (0..15) holds
// K = kk + 16*(j>>3) + 8*(l>>4) + (j&7)  -> two contiguous 16B runs.
__device__ __forceinline__ v16bf ld_frag(const __bf16* base, int row, int ldk,
                                         int kk, int hi) {
    FragBF f;
    const __bf16* p = base + row * ldk + kk + 8 * hi;
    f.u[0] = *reinterpret_cast<const u32x4*>(p);
    f.u[1] = *reinterpret_cast<const u32x4*>(p + 16);
    return f.v;
}

__device__ __forceinline__ v8f wmma_bf16(v16bf a, v16bf b, v8f c) {
    return __builtin_amdgcn_wmma_f32_16x16x32_bf16(
        /*neg_a=*/false, a, /*neg_b=*/false, b,
        /*c_mod=*/(short)0, c, /*reuse_a=*/false, /*reuse_b=*/false);
}

// ---------------------------------------------------------------------------
// Kernel 1: fp32 -> bf16 convert
// ---------------------------------------------------------------------------
__global__ __launch_bounds__(256) void cvt_f32_to_bf16(const float* __restrict__ in,
                                                       __bf16* __restrict__ out,
                                                       int n) {
    int i = blockIdx.x * 256 + threadIdx.x;
    int stride = gridDim.x * 256;
    for (; i < n; i += stride) out[i] = (__bf16)in[i];
}

// ---------------------------------------------------------------------------
// Kernel 2/4: tiled bf16 WMMA GEMM  out[r,o] = sum_k A[r,k] * W[o,k]
// Block tile 128x128, 8 waves (4x2), wave tile 32x64 (2x4 of 16x16).
// K staged 64 at a time; double-buffered LDS filled by async DMA (ASYNCcnt):
// per k-step each wave issues 8 GLOBAL_LOAD_ASYNC_TO_LDS_B128 into the next
// buffer, then waits asynccnt<=8 so the *current* buffer (in-order) is ready.
// mode 0: store f32 to outF [rows, OC]
// mode 1: QKV scatter: o=(s,h,d); fold QK_SCALE into q; store bf16 [B,H,N,D]
// ---------------------------------------------------------------------------
__global__ __launch_bounds__(256) void gemm_bf16_wmma(
    const __bf16* __restrict__ A, const __bf16* __restrict__ W,
    float* __restrict__ outF,
    __bf16* __restrict__ outQ, __bf16* __restrict__ outK, __bf16* __restrict__ outV,
    int K, int OC, int mode)
{
    __shared__ __bf16 As[2][128 * 64];
    __shared__ __bf16 Ws[2][128 * 64];

    const int t    = threadIdx.x;
    const int lane = t & 31;
    const int wave = t >> 5;
    const int wm   = wave >> 1;          // 0..3  (row group of 32)
    const int wn   = wave & 1;           // 0..1  (col group of 64)
    const int hi   = lane >> 4;
    const int ln   = lane & 15;
    const int r0   = blockIdx.y * 128;
    const int c0   = blockIdx.x * 128;

    const v8f vzero = {0.f, 0.f, 0.f, 0.f, 0.f, 0.f, 0.f, 0.f};
    v8f acc[2][4];
    for (int mt = 0; mt < 2; ++mt)
        for (int nt = 0; nt < 4; ++nt) acc[mt][nt] = vzero;

    auto stage = [&](int buf, int k0) {
        for (int c = t; c < 128 * 8; c += 256) {
            int row = c >> 3;
            int c8  = (c & 7) * 8;
#if HAVE_ASYNC
            __builtin_amdgcn_global_load_async_to_lds_b128(
                (AS1 i32x4*)(A + (size_t)(r0 + row) * K + k0 + c8),
                (AS3 i32x4*)&As[buf][row * 64 + c8], 0, 0);
            __builtin_amdgcn_global_load_async_to_lds_b128(
                (AS1 i32x4*)(W + (size_t)(c0 + row) * K + k0 + c8),
                (AS3 i32x4*)&Ws[buf][row * 64 + c8], 0, 0);
#else
            *reinterpret_cast<u32x4*>(&As[buf][row * 64 + c8]) =
                *reinterpret_cast<const u32x4*>(&A[(size_t)(r0 + row) * K + k0 + c8]);
            *reinterpret_cast<u32x4*>(&Ws[buf][row * 64 + c8]) =
                *reinterpret_cast<const u32x4*>(&W[(size_t)(c0 + row) * K + k0 + c8]);
#endif
        }
    };

    stage(0, 0);
    for (int k0 = 0; k0 < K; k0 += 64) {
        const int cur = (k0 >> 6) & 1;
        const bool has_next = (k0 + 64) < K;
        if (has_next) stage(cur ^ 1, k0 + 64);   // DMA overlaps compute below
#if HAVE_ASYNC
        if (has_next) WAIT_ASYNC(8); else WAIT_ASYNC(0);
#endif
        __syncthreads();                          // current buffer visible to all

        const __bf16* Ab = &As[cur][0];
        const __bf16* Wb = &Ws[cur][0];
        for (int kk = 0; kk < 64; kk += 32) {
            v16bf a0 = ld_frag(Ab, wm * 32 + 0  + ln, 64, kk, hi);
            v16bf a1 = ld_frag(Ab, wm * 32 + 16 + ln, 64, kk, hi);
            for (int nt = 0; nt < 4; ++nt) {
                v16bf b = ld_frag(Wb, wn * 64 + nt * 16 + ln, 64, kk, hi);
                acc[0][nt] = wmma_bf16(a0, b, acc[0][nt]);
                acc[1][nt] = wmma_bf16(a1, b, acc[1][nt]);
            }
        }
        __syncthreads();                          // done reading before overwrite
    }

    // epilogue: C/D fragment layout (lane 0-15: M=rr, lane 16-31: M=rr+8)
    for (int mt = 0; mt < 2; ++mt) {
        for (int nt = 0; nt < 4; ++nt) {
            for (int rr = 0; rr < 8; ++rr) {
                int row = r0 + wm * 32 + mt * 16 + rr + 8 * hi;
                int o   = c0 + wn * 64 + nt * 16 + ln;
                float val = acc[mt][nt][rr];
                if (mode == 0) {
                    outF[(size_t)row * OC + o] = val;
                } else {
                    int s = o >> 10;            // 0=q 1=k 2=v
                    int h = (o >> 6) & 15;
                    int d = o & 63;
                    int b = row >> 11;
                    int n = row & 2047;
                    if (s == 0) val *= QK_SCALE;
                    __bf16* dst = (s == 0) ? outQ : ((s == 1) ? outK : outV);
                    dst[((size_t)((b * HEADS + h) * SEQ_N + n)) * HDIM + d] = (__bf16)val;
                }
            }
        }
    }
}

// ---------------------------------------------------------------------------
// Kernel 3: flash attention. grid = (N/64 q-tiles, B*H). block = 128 (4 waves).
// Each wave owns 16 Q rows; loop over 64-wide K/V tiles with online softmax.
// K tile (64x64 bf16, 8KB) is DMA'd into LDS by the Tensor Data Mover from
// wave 0 (one TENSOR_LOAD_TO_LDS per iteration, TENSORcnt). V tile is staged
// transposed by hand (TDM cannot transpose).
// ---------------------------------------------------------------------------
__global__ __launch_bounds__(128) void flash_attn_wmma(
    const __bf16* __restrict__ Q, const __bf16* __restrict__ Kv,
    const __bf16* __restrict__ V, __bf16* __restrict__ O)
{
    __shared__ __bf16 Ks[64 * 64];       // [j][d]
    __shared__ __bf16 VTs[64 * 64];      // [d][j]
    __shared__ __bf16 Ps[4][16 * 64];    // per-wave P scratch [i][j]

    const int t    = threadIdx.x;
    const int lane = t & 31;
    const int wave = t >> 5;
    const int hi   = lane >> 4;
    const int ln   = lane & 15;
    const int q0   = blockIdx.x * 64;
    const int bh   = blockIdx.y;
    const int b    = bh >> 4;
    const int h    = bh & 15;

    const __bf16* qh = Q  + (size_t)bh * SEQ_N * HDIM;
    const __bf16* kh = Kv + (size_t)bh * SEQ_N * HDIM;
    const __bf16* vh = V  + (size_t)bh * SEQ_N * HDIM;

#if HAVE_TDM
    // TDM descriptor, group1..3 are loop-invariant (2D tile 64x64, 2B elems).
    // Bitfields per CDNA5 ISA.
    i32x8 tg1;
    tg1[0] = (1 << 16);          // workgroup_mask=0, data_size=1 (2 bytes)
    tg1[1] = (HDIM << 16);       // tensor_dim0[15:0] in bits 63:48
    tg1[2] = (SEQ_N & 0xFFFF) << 16; // tensor_dim0 hi=0, tensor_dim1 lo16
    tg1[3] = (64 << 16);         // tensor_dim1 hi=0, tile_dim0=64
    tg1[4] = 64;                 // tile_dim1=64, tile_dim2=0
    tg1[5] = HDIM;               // tensor_dim0_stride = 64 (lo 32)
    tg1[6] = 0;                  // stride hi, tensor_dim1_stride lo (unused)
    tg1[7] = 0;
    i32x4 tzero4 = {0, 0, 0, 0};
    const unsigned int ks_lds = (unsigned int)(size_t)(void*)&Ks[0];
#endif

    // Q fragments stay in registers for the whole kernel (16 rows x 64 d)
    FragBF qf[2];
    {
        const int qrow = q0 + wave * 16 + ln;
        for (int kk = 0; kk < 2; ++kk) {
            const __bf16* p = qh + (size_t)qrow * HDIM + kk * 32 + 8 * hi;
            qf[kk].u[0] = *reinterpret_cast<const u32x4*>(p);
            qf[kk].u[1] = *reinterpret_cast<const u32x4*>(p + 16);
        }
    }

    const v8f vzero = {0.f, 0.f, 0.f, 0.f, 0.f, 0.f, 0.f, 0.f};
    float rmax[8], rsum[8];
    v8f oacc[4];
    for (int r = 0; r < 8; ++r) { rmax[r] = -1e30f; rsum[r] = 0.f; }
    for (int dt = 0; dt < 4; ++dt) oacc[dt] = vzero;

    for (int j0 = 0; j0 < SEQ_N; j0 += 64) {
        __syncthreads();   // everyone done with previous K/V tiles

#if HAVE_TDM
        if (wave == 0) {
            unsigned long long ga =
                (unsigned long long)(const void*)(kh + (size_t)j0 * HDIM);
            u32x4 tg0;
            tg0[0] = 1u;                                   // count=1, user mode
            tg0[1] = ks_lds;                               // lds_addr
            tg0[2] = (unsigned int)(ga & 0xFFFFFFFFull);   // global_addr lo
            tg0[3] = (unsigned int)((ga >> 32) & 0x1FFFFFFull) | (2u << 30); // hi + type=2
#if __has_include(<hip/amd_detail/amd_gfx1250_TDM.h>)
            i32x8 z8 = {0, 0, 0, 0, 0, 0, 0, 0};
            __builtin_amdgcn_tensor_load_to_lds(tg0, tg1, tzero4, tzero4, z8, 0);
#else
            __builtin_amdgcn_tensor_load_to_lds(tg0, tg1, tzero4, tzero4, 0);
#endif
        }
#endif
        // stage transposed V tile [d][j] (and K tile if no TDM path)
        for (int c = t; c < 64 * 8; c += 128) {
            int row = c >> 3;
            int c8  = (c & 7) * 8;
#if !HAVE_TDM
            *reinterpret_cast<u32x4*>(&Ks[row * 64 + c8]) =
                *reinterpret_cast<const u32x4*>(&kh[(size_t)(j0 + row) * HDIM + c8]);
#endif
            union { u32x4 u; __bf16 e[8]; } vv;
            vv.u = *reinterpret_cast<const u32x4*>(&vh[(size_t)(j0 + row) * HDIM + c8]);
            for (int e = 0; e < 8; ++e) VTs[(c8 + e) * 64 + row] = vv.e[e];
        }
#if HAVE_TDM
        if (wave == 0) __builtin_amdgcn_s_wait_tensorcnt(0);
#endif
        __syncthreads();

        // S = Q * K^T  (16 x 64), accumulate over d=64 in two k-steps
        v8f sacc[4];
        for (int nt = 0; nt < 4; ++nt) sacc[nt] = vzero;
        for (int nt = 0; nt < 4; ++nt)
            for (int kk = 0; kk < 2; ++kk) {
                v16bf bf = ld_frag(Ks, nt * 16 + ln, 64, kk * 32, hi);
                sacc[nt] = wmma_bf16(qf[kk].v, bf, sacc[nt]);
            }

        // online softmax: rows rr+8*hi live across the 16-lane half-wave
        float tmax[8], alpha[8], tsum[8];
        for (int r = 0; r < 8; ++r)
            tmax[r] = fmaxf(fmaxf(sacc[0][r], sacc[1][r]),
                            fmaxf(sacc[2][r], sacc[3][r]));
        for (int off = 1; off < 16; off <<= 1)
            for (int r = 0; r < 8; ++r)
                tmax[r] = fmaxf(tmax[r], __shfl_xor(tmax[r], off, 32));
        for (int r = 0; r < 8; ++r) {
            float nm = fmaxf(rmax[r], tmax[r]);
            alpha[r] = __expf(rmax[r] - nm);
            rmax[r]  = nm;
            tsum[r]  = 0.f;
        }
        for (int nt = 0; nt < 4; ++nt)
            for (int r = 0; r < 8; ++r) {
                float p = __expf(sacc[nt][r] - rmax[r]);
                sacc[nt][r] = p;
                tsum[r] += p;
            }
        for (int off = 1; off < 16; off <<= 1)
            for (int r = 0; r < 8; ++r)
                tsum[r] += __shfl_xor(tsum[r], off, 32);
        for (int r = 0; r < 8; ++r) rsum[r] = rsum[r] * alpha[r] + tsum[r];
        for (int dt = 0; dt < 4; ++dt)
            for (int r = 0; r < 8; ++r) oacc[dt][r] *= alpha[r];

        // P (C/D layout) -> per-wave LDS -> A-fragment layout
        __bf16* pw = &Ps[wave][0];
        for (int nt = 0; nt < 4; ++nt)
            for (int r = 0; r < 8; ++r)
                pw[(r + 8 * hi) * 64 + nt * 16 + ln] = (__bf16)sacc[nt][r];
        // same-wave LDS store->load ordering
        asm volatile("s_wait_dscnt 0" ::: "memory");

        // O += P * V  (16 x 64)
        for (int kk = 0; kk < 2; ++kk) {
            v16bf pf = ld_frag(pw, ln, 64, kk * 32, hi);
            for (int dt = 0; dt < 4; ++dt) {
                v16bf vf = ld_frag(VTs, dt * 16 + ln, 64, kk * 32, hi);
                oacc[dt] = wmma_bf16(pf, vf, oacc[dt]);
            }
        }
    }

    // normalize + store to [B*N, C] (col = h*64 + d)
    for (int dt = 0; dt < 4; ++dt)
        for (int r = 0; r < 8; ++r) {
            int row = q0 + wave * 16 + r + 8 * hi;
            int col = h * HDIM + dt * 16 + ln;
            float val = oacc[dt][r] / rsum[r];
            O[(size_t)(b * SEQ_N + row) * EMB_C + col] = (__bf16)val;
        }
}

// ---------------------------------------------------------------------------
// Launcher (workspace: 48 MB of bf16 staging in d_ws)
// ---------------------------------------------------------------------------
extern "C" void kernel_launch(void* const* d_in, const int* in_sizes, int n_in,
                              void* d_out, int out_size, void* d_ws, size_t ws_size,
                              hipStream_t stream) {
    (void)in_sizes; (void)n_in; (void)out_size; (void)ws_size;
    const float* x      = (const float*)d_in[0];   // [2,2048,1024]
    const float* w_qkv  = (const float*)d_in[1];   // [3072,1024]
    const float* w_proj = (const float*)d_in[2];   // [1024,1024]
    float* out = (float*)d_out;                    // [2,2048,1024]

    const size_t MB = 1024ull * 1024ull;
    char* ws = (char*)d_ws;
    __bf16* xb     = (__bf16*)(ws + 0 * MB);   // 8 MB  [4096,1024]
    __bf16* wqkvb  = (__bf16*)(ws + 8 * MB);   // 6 MB  [3072,1024]
    __bf16* wprojb = (__bf16*)(ws + 14 * MB);  // 2 MB  [1024,1024]
    __bf16* qb     = (__bf16*)(ws + 16 * MB);  // 8 MB  [B,H,N,D] (pre-scaled)
    __bf16* kb     = (__bf16*)(ws + 24 * MB);  // 8 MB
    __bf16* vb     = (__bf16*)(ws + 32 * MB);  // 8 MB
    __bf16* attnb  = (__bf16*)(ws + 40 * MB);  // 8 MB  [4096,1024]

    const int nx = BATCH * SEQ_N * EMB_C;          // 4194304
    const int nq = 3 * EMB_C * EMB_C;              // 3145728
    const int np = EMB_C * EMB_C;                  // 1048576
    cvt_f32_to_bf16<<<1024, 256, 0, stream>>>(x, xb, nx);
    cvt_f32_to_bf16<<<1024, 256, 0, stream>>>(w_qkv, wqkvb, nq);
    cvt_f32_to_bf16<<<1024, 256, 0, stream>>>(w_proj, wprojb, np);

    // QKV projection: [4096,1024] x [3072,1024]^T, scatter to q/k/v
    gemm_bf16_wmma<<<dim3(3 * EMB_C / 128, BATCH * SEQ_N / 128), 256, 0, stream>>>(
        xb, wqkvb, nullptr, qb, kb, vb, EMB_C, 3 * EMB_C, /*mode=*/1);

    // attention per (b,h), 64-row q tiles
    flash_attn_wmma<<<dim3(SEQ_N / 64, BATCH * HEADS), 128, 0, stream>>>(
        qb, kb, vb, attnb);

    // output projection: [4096,1024] x [1024,1024]^T -> fp32 out
    gemm_bf16_wmma<<<dim3(EMB_C / 128, BATCH * SEQ_N / 128), 256, 0, stream>>>(
        attnb, wprojb, out, nullptr, nullptr, nullptr, EMB_C, EMB_C, /*mode=*/0);
}
